// GCNConvNet_25323127177385
// MI455X (gfx1250) — matrix operand order repaired and checked by reading, not compiled, not run on hardware
//
#include <hip/hip_runtime.h>
#include <math.h>

typedef __attribute__((ext_vector_type(16))) _Float16 v16h;
typedef __attribute__((ext_vector_type(8)))  float    v8f;

#define SLOPE 0.2f
#define EPS_SM 1e-16f
#define BN_EPS 1e-5f

// ---- order-preserving float<->uint encoding for atomic max on f32 ----
__device__ __forceinline__ unsigned fenc(float f) {
    unsigned u = __float_as_uint(f);
    return (u & 0x80000000u) ? ~u : (u | 0x80000000u);
}
__device__ __forceinline__ float fdec(unsigned u) {
    unsigned v = (u & 0x80000000u) ? (u & 0x7FFFFFFFu) : ~u;
    return __uint_as_float(v);
}
__device__ __forceinline__ float lrelu(float v) { return v > 0.f ? v : SLOPE * v; }
__device__ __forceinline__ float relu(float v)  { return v > 0.f ? v : 0.f; }

// ================= GAT layer 1 (H=1, C=3) node projection =================
__global__ void gat_node1(const float* __restrict__ x, const float* __restrict__ W1,
                          const float* __restrict__ as1, const float* __restrict__ ad1,
                          float* __restrict__ h1, float* __restrict__ es1, float* __restrict__ ed1,
                          unsigned* __restrict__ m1, float* __restrict__ s1, float* __restrict__ out1,
                          int N) {
    int n = blockIdx.x * blockDim.x + threadIdx.x;
    if (n >= N) return;
    float x0 = x[n*3+0], x1 = x[n*3+1], x2 = x[n*3+2];
    float h[3], es = 0.f, ed = 0.f;
    #pragma unroll
    for (int c = 0; c < 3; ++c) {
        h[c] = x0 * W1[c] + x1 * W1[3+c] + x2 * W1[6+c];
        es += h[c] * as1[c];
        ed += h[c] * ad1[c];
        h1[n*3+c] = h[c];
        out1[n*3+c] = 0.f;
    }
    es1[n] = es; ed1[n] = ed;
    m1[n] = fenc(-INFINITY);
    s1[n] = 0.f;
}

// ================= edge passes, H=1 =================
__global__ void edge1_max(const int* __restrict__ ei, int E, int N,
                          const float* __restrict__ es, const float* __restrict__ ed,
                          unsigned* __restrict__ m) {
    int e = blockIdx.x * blockDim.x + threadIdx.x;
    if (e >= E + N) return;
    int s, d;
    if (e < E) { s = ei[e]; d = ei[E + e]; } else { s = d = e - E; }
    atomicMax(&m[d], fenc(lrelu(es[s] + ed[d])));
}
__global__ void edge1_sum(const int* __restrict__ ei, int E, int N,
                          const float* __restrict__ es, const float* __restrict__ ed,
                          const unsigned* __restrict__ m, float* __restrict__ ss) {
    int e = blockIdx.x * blockDim.x + threadIdx.x;
    if (e >= E + N) return;
    int s, d;
    if (e < E) { s = ei[e]; d = ei[E + e]; } else { s = d = e - E; }
    float v = lrelu(es[s] + ed[d]);
    atomicAdd(&ss[d], __expf(v - fdec(m[d])));
}
__global__ void edge1_msg(const int* __restrict__ ei, int E, int N,
                          const float* __restrict__ es, const float* __restrict__ ed,
                          const unsigned* __restrict__ m, const float* __restrict__ ss,
                          const float* __restrict__ h, float* __restrict__ out) {
    int e = blockIdx.x * blockDim.x + threadIdx.x;
    if (e >= E + N) return;
    int s, d;
    if (e < E) { s = ei[e]; d = ei[E + e]; } else { s = d = e - E; }
    float v = lrelu(es[s] + ed[d]);
    float alpha = __expf(v - fdec(m[d])) / (ss[d] + EPS_SM);
    #pragma unroll
    for (int c = 0; c < 3; ++c)
        atomicAdd(&out[d*3+c], h[s*3+c] * alpha);
}

// ================= GAT layer 2 (H=4, C=3) node projection =================
__global__ void gat_node2(const float* __restrict__ out1, const float* __restrict__ b1,
                          const float* __restrict__ W2, const float* __restrict__ as2,
                          const float* __restrict__ ad2,
                          float* __restrict__ h2, float* __restrict__ es2, float* __restrict__ ed2,
                          unsigned* __restrict__ m2, float* __restrict__ s2, float* __restrict__ out2,
                          int N) {
    int n = blockIdx.x * blockDim.x + threadIdx.x;
    if (n >= N) return;
    float xv[3];
    #pragma unroll
    for (int c = 0; c < 3; ++c) xv[c] = relu(out1[n*3+c] + b1[c]);
    float h[12];
    #pragma unroll
    for (int j = 0; j < 12; ++j) {
        h[j] = xv[0] * W2[j] + xv[1] * W2[12+j] + xv[2] * W2[24+j];
        h2[n*12+j] = h[j];
        out2[n*12+j] = 0.f;
    }
    #pragma unroll
    for (int hd = 0; hd < 4; ++hd) {
        float es = 0.f, ed = 0.f;
        #pragma unroll
        for (int c = 0; c < 3; ++c) {
            es += h[hd*3+c] * as2[hd*3+c];
            ed += h[hd*3+c] * ad2[hd*3+c];
        }
        es2[n*4+hd] = es; ed2[n*4+hd] = ed;
        m2[n*4+hd] = fenc(-INFINITY);
        s2[n*4+hd] = 0.f;
    }
}

// ================= edge passes, H=4 (float4-vectorized node gathers) =====
__global__ void edge4_max(const int* __restrict__ ei, int E, int N,
                          const float* __restrict__ es, const float* __restrict__ ed,
                          unsigned* __restrict__ m) {
    int e = blockIdx.x * blockDim.x + threadIdx.x;
    if (e >= E + N) return;
    int s, d;
    if (e < E) { s = ei[e]; d = ei[E + e]; } else { s = d = e - E; }
    float4 a = ((const float4*)es)[s];
    float4 b = ((const float4*)ed)[d];
    atomicMax(&m[d*4+0], fenc(lrelu(a.x + b.x)));
    atomicMax(&m[d*4+1], fenc(lrelu(a.y + b.y)));
    atomicMax(&m[d*4+2], fenc(lrelu(a.z + b.z)));
    atomicMax(&m[d*4+3], fenc(lrelu(a.w + b.w)));
}
__global__ void edge4_sum(const int* __restrict__ ei, int E, int N,
                          const float* __restrict__ es, const float* __restrict__ ed,
                          const unsigned* __restrict__ m, float* __restrict__ ss) {
    int e = blockIdx.x * blockDim.x + threadIdx.x;
    if (e >= E + N) return;
    int s, d;
    if (e < E) { s = ei[e]; d = ei[E + e]; } else { s = d = e - E; }
    float4 a = ((const float4*)es)[s];
    float4 b = ((const float4*)ed)[d];
    uint4  mv = ((const uint4*)m)[d];
    atomicAdd(&ss[d*4+0], __expf(lrelu(a.x + b.x) - fdec(mv.x)));
    atomicAdd(&ss[d*4+1], __expf(lrelu(a.y + b.y) - fdec(mv.y)));
    atomicAdd(&ss[d*4+2], __expf(lrelu(a.z + b.z) - fdec(mv.z)));
    atomicAdd(&ss[d*4+3], __expf(lrelu(a.w + b.w) - fdec(mv.w)));
}
__global__ void edge4_msg(const int* __restrict__ ei, int E, int N,
                          const float* __restrict__ es, const float* __restrict__ ed,
                          const unsigned* __restrict__ m, const float* __restrict__ ss,
                          const float* __restrict__ h, float* __restrict__ out) {
    int e = blockIdx.x * blockDim.x + threadIdx.x;
    if (e >= E + N) return;
    int s, d;
    if (e < E) { s = ei[e]; d = ei[E + e]; } else { s = d = e - E; }
    float4 a  = ((const float4*)es)[s];
    float4 b  = ((const float4*)ed)[d];
    uint4  mv = ((const uint4*)m)[d];
    float4 sv = ((const float4*)ss)[d];
    float alpha[4];
    alpha[0] = __expf(lrelu(a.x + b.x) - fdec(mv.x)) / (sv.x + EPS_SM);
    alpha[1] = __expf(lrelu(a.y + b.y) - fdec(mv.y)) / (sv.y + EPS_SM);
    alpha[2] = __expf(lrelu(a.z + b.z) - fdec(mv.z)) / (sv.z + EPS_SM);
    alpha[3] = __expf(lrelu(a.w + b.w) - fdec(mv.w)) / (sv.w + EPS_SM);
    const float4* hs = (const float4*)(h + (size_t)s * 12);
    float4 h0 = hs[0], h1 = hs[1], h2v = hs[2];
    float hv[12] = {h0.x,h0.y,h0.z,h0.w, h1.x,h1.y,h1.z,h1.w, h2v.x,h2v.y,h2v.z,h2v.w};
    #pragma unroll
    for (int hd = 0; hd < 4; ++hd)
        #pragma unroll
        for (int c = 0; c < 3; ++c)
            atomicAdd(&out[d*12 + hd*3 + c], hv[hd*3+c] * alpha[hd]);
}

// ======= FC layer via WMMA: relu(out2 + b2) [N,12] @ fc_W[12,128] =========
// A fragment f16 layout (16x32, K padded 12->32), per CDNA5 ISA 7.12.2:
//   group0 lanes: half t -> K = t (t<8) / 8+t (t>=8)
//   group1 lanes: half t -> K = 8+t (t<8) / 16+t (t>=8)
// B fragment: same K mapping as A, N = lane&15.
__global__ __launch_bounds__(256) void fc_wmma(const float* __restrict__ out2,
                          const float* __restrict__ b2,
                          const float* __restrict__ fcW, const float* __restrict__ fcb,
                          float* __restrict__ h128, int N) {
    int row0 = blockIdx.x * 16;
    int lane = threadIdx.x;            // 0..31
    int ct   = threadIdx.y;            // col tile 0..7
    int grp  = lane >> 4;
    int l    = lane & 15;
    int col  = ct * 16 + l;
    int arow = row0 + l;
    bool fullTile = (row0 + 16 <= N);  // wave-uniform

    v16h a, b;
    if (fullTile) {
        #pragma unroll
        for (int t = 0; t < 16; ++t) {
            int K = t + ((t < 8) ? (grp ? 8 : 0) : (grp ? 16 : 8));
            float v = (K < 12) ? relu(out2[arow*12 + K] + b2[K]) : 0.f;
            a[t] = (_Float16)v;
        }
    } else {
        bool rok = arow < N;
        #pragma unroll
        for (int t = 0; t < 16; ++t) {
            int K = t + ((t < 8) ? (grp ? 8 : 0) : (grp ? 16 : 8));
            float v = 0.f;
            if (rok && K < 12) v = relu(out2[arow*12 + K] + b2[K]);
            a[t] = (_Float16)v;
        }
    }
    #pragma unroll
    for (int t = 0; t < 16; ++t) {
        int K = t + ((t < 8) ? (grp ? 8 : 0) : (grp ? 16 : 8));
        b[t] = (K < 12) ? (_Float16)fcW[K*128 + col] : (_Float16)0.f;
    }
    v8f c = {};
    c = __builtin_amdgcn_wmma_f32_16x16x32_f16(false, a, false, b, (short)0, c, false, false);
    float bias = fcb[col];
    if (fullTile) {
        #pragma unroll
        for (int i = 0; i < 8; ++i) {
            int r = row0 + i + (grp ? 8 : 0);
            h128[(size_t)r*128 + col] = relu(c[i] + bias);
        }
    } else {
        #pragma unroll
        for (int i = 0; i < 8; ++i) {
            int r = row0 + i + (grp ? 8 : 0);
            if (r < N) h128[(size_t)r*128 + col] = relu(c[i] + bias);
        }
    }
}

// ================= BatchNorm statistics =================
__global__ void bn_stats(const float* __restrict__ h128, float* __restrict__ bsum,
                         float* __restrict__ bsq, int N) {
    int c = threadIdx.x; // 0..127
    float su = 0.f, sq = 0.f;
    for (int r = blockIdx.x; r < N; r += gridDim.x) {
        float v = h128[(size_t)r*128 + c];
        su += v; sq += v * v;
    }
    atomicAdd(&bsum[c], su);
    atomicAdd(&bsq[c], sq);
}
__global__ void bn_finalize(const float* __restrict__ bsum, const float* __restrict__ bsq,
                            const float* __restrict__ g, const float* __restrict__ bb,
                            float* __restrict__ scale, float* __restrict__ shift, int N) {
    int c = threadIdx.x;
    float fN = (float)N;
    float mu = bsum[c] / fN;
    float var = bsq[c] / fN - mu * mu;
    float sc = g[c] * rsqrtf(var + BN_EPS);
    scale[c] = sc;
    shift[c] = bb[c] - mu * sc;
}

// ==== fused: BN-apply -> [N,128]@l2_W[128,64]+b (WMMA) -> @out_W[64,6]+b -> sigmoid ====
__global__ __launch_bounds__(128) void l2_out_fused(const float* __restrict__ h128,
                             const float* __restrict__ scale, const float* __restrict__ shift,
                             const float* __restrict__ l2W, const float* __restrict__ l2b,
                             const float* __restrict__ outW, const float* __restrict__ outb,
                             float* __restrict__ out, int N) {
    __shared__ float tile[16][65];
    int row0 = blockIdx.x * 16;
    int lane = threadIdx.x;            // 0..31
    int ct   = threadIdx.y;            // col tile 0..3
    int grp  = lane >> 4;
    int l    = lane & 15;
    int col  = ct * 16 + l;
    int arow = row0 + l;
    bool fullTile = (row0 + 16 <= N);  // wave-uniform
    bool rok = arow < N;

    v8f c = {};
    if (fullTile) {
        #pragma unroll
        for (int ko = 0; ko < 128; ko += 32) {
            v16h a, b;
            #pragma unroll
            for (int t = 0; t < 16; ++t) {
                int K = ko + t + ((t < 8) ? (grp ? 8 : 0) : (grp ? 16 : 8));
                a[t] = (_Float16)(h128[(size_t)arow*128 + K] * scale[K] + shift[K]);
            }
            #pragma unroll
            for (int t = 0; t < 16; ++t) {
                int K = ko + t + ((t < 8) ? (grp ? 8 : 0) : (grp ? 16 : 8));
                b[t] = (_Float16)l2W[K*64 + col];
            }
            c = __builtin_amdgcn_wmma_f32_16x16x32_f16(false, a, false, b, (short)0, c, false, false);
        }
    } else {
        #pragma unroll
        for (int ko = 0; ko < 128; ko += 32) {
            v16h a, b;
            #pragma unroll
            for (int t = 0; t < 16; ++t) {
                int K = ko + t + ((t < 8) ? (grp ? 8 : 0) : (grp ? 16 : 8));
                float v = 0.f;
                if (rok) v = h128[(size_t)arow*128 + K] * scale[K] + shift[K];
                a[t] = (_Float16)v;
            }
            #pragma unroll
            for (int t = 0; t < 16; ++t) {
                int K = ko + t + ((t < 8) ? (grp ? 8 : 0) : (grp ? 16 : 8));
                b[t] = (_Float16)l2W[K*64 + col];
            }
            c = __builtin_amdgcn_wmma_f32_16x16x32_f16(false, a, false, b, (short)0, c, false, false);
        }
    }
    float bias = l2b[col];
    #pragma unroll
    for (int i = 0; i < 8; ++i)
        tile[i + (grp ? 8 : 0)][col] = c[i] + bias;
    __syncthreads();

    int t = threadIdx.y * 32 + threadIdx.x;  // 0..127
    if (t < 96) {
        int r = t / 6, j = t % 6;
        int grow = row0 + r;
        if (grow < N) {
            float acc = outb[j];
            #pragma unroll
            for (int k = 0; k < 64; ++k) acc += tile[r][k] * outW[k*6 + j];
            out[(size_t)grow*6 + j] = 1.f / (1.f + __expf(-acc));
        }
    }
}

extern "C" void kernel_launch(void* const* d_in, const int* in_sizes, int n_in,
                              void* d_out, int out_size, void* d_ws, size_t ws_size,
                              hipStream_t stream) {
    const float* x    = (const float*)d_in[0];
    const int*   ei   = (const int*)  d_in[1];
    const float* W1   = (const float*)d_in[2];
    const float* as1  = (const float*)d_in[3];
    const float* ad1  = (const float*)d_in[4];
    const float* b1   = (const float*)d_in[5];
    const float* W2   = (const float*)d_in[6];
    const float* as2  = (const float*)d_in[7];
    const float* ad2  = (const float*)d_in[8];
    const float* b2   = (const float*)d_in[9];
    const float* fcW  = (const float*)d_in[10];
    const float* fcb  = (const float*)d_in[11];
    const float* bng  = (const float*)d_in[12];
    const float* bnb  = (const float*)d_in[13];
    const float* l2W  = (const float*)d_in[14];
    const float* l2b  = (const float*)d_in[15];
    const float* outW = (const float*)d_in[16];
    const float* outb = (const float*)d_in[17];
    float* out = (float*)d_out;

    int N = in_sizes[0] / 3;
    int E = in_sizes[1] / 2;

    // workspace layout (floats)
    float* w = (float*)d_ws;
    size_t o = 0;
    float*    h1   = w + o; o += (size_t)N * 3;
    float*    es1  = w + o; o += (size_t)N;
    float*    ed1  = w + o; o += (size_t)N;
    unsigned* m1   = (unsigned*)(w + o); o += (size_t)N;
    float*    s1   = w + o; o += (size_t)N;
    float*    out1 = w + o; o += (size_t)N * 3;
    float*    h2   = w + o; o += (size_t)N * 12;
    float*    es2  = w + o; o += (size_t)N * 4;
    float*    ed2  = w + o; o += (size_t)N * 4;
    unsigned* m2   = (unsigned*)(w + o); o += (size_t)N * 4;
    float*    s2   = w + o; o += (size_t)N * 4;
    float*    out2 = w + o; o += (size_t)N * 12;
    float*    h128 = w + o; o += (size_t)N * 128;
    float*    bsum = w + o; o += 128;
    float*    bsq  = w + o; o += 128;
    float*    bsc  = w + o; o += 128;
    float*    bsh  = w + o; o += 128;

    int nodeBlocks = (N + 255) / 256;
    int edgeBlocks = (E + N + 255) / 256;
    int rowTiles   = (N + 15) / 16;

    // ---- GAT layer 1 ----
    gat_node1<<<nodeBlocks, 256, 0, stream>>>(x, W1, as1, ad1, h1, es1, ed1, m1, s1, out1, N);
    edge1_max<<<edgeBlocks, 256, 0, stream>>>(ei, E, N, es1, ed1, m1);
    edge1_sum<<<edgeBlocks, 256, 0, stream>>>(ei, E, N, es1, ed1, m1, s1);
    edge1_msg<<<edgeBlocks, 256, 0, stream>>>(ei, E, N, es1, ed1, m1, s1, h1, out1);

    // ---- GAT layer 2 ----
    gat_node2<<<nodeBlocks, 256, 0, stream>>>(out1, b1, W2, as2, ad2, h2, es2, ed2, m2, s2, out2, N);
    edge4_max<<<edgeBlocks, 256, 0, stream>>>(ei, E, N, es2, ed2, m2);
    edge4_sum<<<edgeBlocks, 256, 0, stream>>>(ei, E, N, es2, ed2, m2, s2);
    edge4_msg<<<edgeBlocks, 256, 0, stream>>>(ei, E, N, es2, ed2, m2, s2, h2, out2);

    // ---- dense head: FC (WMMA) ----
    fc_wmma<<<dim3(rowTiles), dim3(32, 8), 0, stream>>>(out2, b2, fcW, fcb, h128, N);

    // ---- BN stats ----
    hipMemsetAsync(bsum, 0, 2 * 128 * sizeof(float), stream);
    int statBlocks = 1024 < N ? 1024 : N;
    bn_stats<<<statBlocks, 128, 0, stream>>>(h128, bsum, bsq, N);
    bn_finalize<<<1, 128, 0, stream>>>(bsum, bsq, bng, bnb, bsc, bsh, N);

    // ---- fused BN-apply + L2 (WMMA) + out layer + sigmoid ----
    l2_out_fused<<<dim3(rowTiles), dim3(32, 4), 0, stream>>>(h128, bsc, bsh, l2W, l2b,
                                                             outW, outb, out, N);
    (void)ws_size; (void)n_in; (void)out_size;
}
